// OwnRefTransformerActor_69020124447505
// MI455X (gfx1250) — matrix operand order repaired and checked by reading, not compiled, not run on hardware
//
#include <hip/hip_runtime.h>
#include <hip/hip_bf16.h>
#include <math.h>

// ---------------------------------------------------------------------------
// Problem constants (from reference)
// ---------------------------------------------------------------------------
#define BB      1024
#define TT      64
#define INDIM   8
#define OUTDIM  2
#define HH      3
#define KK      5
#define HID     256
#define NTOK    (BB * TT)          // 65536 tokens
#define X0K     32                 // 23 padded to 32 for one 16x16x32 K-step

typedef __attribute__((ext_vector_type(16))) _Float16 v16h;
typedef __attribute__((ext_vector_type(8)))  _Float16 v8h;
typedef __attribute__((ext_vector_type(8)))  float    v8f;

// ---------------------------------------------------------------------------
// Kernel 0: pack W1 (23x256, zero-pad K->32) and W2 (256x256) into f16
// B-fragment order used here (contiguous k): [ntile][kstep][lane][half]
//   column n = ntile*16 + lane%16 ; k = kstep*32 + half + (lane<16 ? 0 : 16)
// ---------------------------------------------------------------------------
__global__ __launch_bounds__(256) void prep_kernel(
    const float* __restrict__ W1, const float* __restrict__ W2,
    _Float16* __restrict__ W1p, _Float16* __restrict__ W2p)
{
    int tid = blockIdx.x * 256 + threadIdx.x;
    if (tid < 16 * 512) {
        // W1p: 16 ntiles x (32 lanes x 16 halves), single K-step
        int ntile = tid >> 9;
        int rem   = tid & 511;
        int lane  = rem >> 4;
        int half  = rem & 15;
        int n = ntile * 16 + (lane & 15);
        int k = half + ((lane < 16) ? 0 : 16);
        float v = (k < 23) ? W1[k * HID + n] : 0.0f;
        W1p[tid] = (_Float16)v;
    } else if (tid < 16 * 512 + 16 * 8 * 512) {
        int u = tid - 16 * 512;
        int ntile = u >> 12;            // /4096
        int rem   = u & 4095;
        int kstep = rem >> 9;           // /512
        int lane  = (rem >> 4) & 31;
        int half  = rem & 15;
        int n = ntile * 16 + (lane & 15);
        int k = kstep * 32 + half + ((lane < 16) ? 0 : 16);
        W2p[u] = (_Float16)W2[k * HID + n];
    }
}

// ---------------------------------------------------------------------------
// Kernel 1: geometry + 3-head/5-dim attention per token.
// One 64-thread block per token; thread j handles neighbor j (j<63).
// Writes x0 = [att(15) | state(8) | 0-pad] as f16, row-major [token][32].
// ---------------------------------------------------------------------------
__global__ __launch_bounds__(64) void attn_kernel(
    const float* __restrict__ state,
    const float* __restrict__ Wq, const float* __restrict__ Wk,
    const float* __restrict__ Wv, _Float16* __restrict__ X0)
{
    const int token = blockIdx.x;
    const int b = token >> 6;
    const int i = token & 63;
    const int j = threadIdx.x;

    __shared__ float sWq[INDIM * 15];
    __shared__ float sWk[5 * 15];
    __shared__ float sWv[5 * 15];
    __shared__ float sSt[TT * INDIM];
    __shared__ float sScore[3][64];
    __shared__ float sMx[3], sSum[3];
    __shared__ float sAtt[16];

    for (int u = j; u < INDIM * 15; u += 64) sWq[u] = Wq[u];
    for (int u = j; u < 5 * 15;     u += 64) { sWk[u] = Wk[u]; sWv[u] = Wv[u]; }
    for (int u = j; u < TT * INDIM; u += 64) sSt[u] = state[(size_t)b * TT * INDIM + u];
    if (j < 15) sAtt[j] = 0.0f;
    __syncthreads();

    float si[INDIM];
#pragma unroll
    for (int c = 0; c < INDIM; ++c) si[c] = sSt[i * INDIM + c];

    // q = state_i @ Wq  (15 outputs)
    float q[15];
#pragma unroll
    for (int o = 0; o < 15; ++o) {
        float a = 0.f;
#pragma unroll
        for (int c = 0; c < INDIM; ++c) a += si[c] * sWq[c * 15 + o];
        q[o] = a;
    }

    const bool active = (j < 63);
    const int jn = (j < i) ? j : j + 1;   // off-diagonal neighbor index
    float sc[3] = {-1e30f, -1e30f, -1e30f};
    float val[15];

    if (active) {
        float d0 = sSt[jn * 8 + 0] - si[0];
        float d1 = sSt[jn * 8 + 1] - si[1];
        float d2 = sSt[jn * 8 + 2] - si[2];
        float d3 = sSt[jn * 8 + 3] - si[3];
        float ang = 1.5707963267948966f - sSt[jn * 8 + 7];
        float cv = __cosf(ang), sv = __sinf(ang);
        float r0 = d0 * cv - d1 * sv;     // xn0
        float r1 = d0 * sv + d1 * cv;     // yn0
        float r2 = d2 * cv - d3 * sv;     // xn1
        float r3 = d2 * sv + d3 * cv;     // yn1
        float r  = sqrtf(r0 * r0 + r1 * r1);
        float rt = 1.0f / (1.0f + __expf(-(1.0f - 5.0f * (r - 0.2f))));
        float reln[5] = { r0 / r, r1 / r, r2, r3, rt };

        float key[15];
#pragma unroll
        for (int o = 0; o < 15; ++o) {
            float ka = 0.f, va = 0.f;
#pragma unroll
            for (int f = 0; f < 5; ++f) {
                ka += reln[f] * sWk[f * 15 + o];
                va += reln[f] * sWv[f * 15 + o];
            }
            key[o] = ka; val[o] = va;
        }
        const float inv_sqrt_k = 0.4472135954999579f;   // 1/sqrt(5)
#pragma unroll
        for (int h = 0; h < 3; ++h) {
            float a = 0.f;
#pragma unroll
            for (int k = 0; k < 5; ++k) a += key[h * 5 + k] * q[h * 5 + k];
            sc[h] = a * inv_sqrt_k;
        }
    }
#pragma unroll
    for (int h = 0; h < 3; ++h) sScore[h][j] = sc[h];
    __syncthreads();

    if (j == 0) {
#pragma unroll
        for (int h = 0; h < 3; ++h) {
            float mx = -1e30f;
            for (int n = 0; n < 63; ++n) mx = fmaxf(mx, sScore[h][n]);
            float sm = 0.f;
            for (int n = 0; n < 63; ++n) sm += __expf(sScore[h][n] - mx);
            sMx[h] = mx; sSum[h] = sm;
        }
    }
    __syncthreads();

    if (active) {
#pragma unroll
        for (int h = 0; h < 3; ++h) {
            float w = __expf(sc[h] - sMx[h]) / sSum[h];
#pragma unroll
            for (int k = 0; k < 5; ++k)
                atomicAdd(&sAtt[h * 5 + k], w * val[h * 5 + k]);
        }
    }
    __syncthreads();

    if (j < X0K) {
        float v = (j < 15) ? sAtt[j] : ((j < 23) ? si[j - 15] : 0.0f);
        X0[(size_t)token * X0K + j] = (_Float16)v;
    }
}

// ---------------------------------------------------------------------------
// Kernel 2: fused MLP 23(->32)->256->256->heads with WMMA f16 (f32 accum).
// Block = 256 threads = 8 wave32, tile = 16 tokens.  Each wave owns 2 N-tiles.
// ---------------------------------------------------------------------------
__global__ __launch_bounds__(256) void mlp_kernel(
    const _Float16* __restrict__ X0,
    const _Float16* __restrict__ W1p, const _Float16* __restrict__ W2p,
    const float* __restrict__ b1, const float* __restrict__ b2,
    const float* __restrict__ Wmu, const float* __restrict__ bmu,
    const float* __restrict__ Wls, const float* __restrict__ bls,
    const float* __restrict__ noise, float* __restrict__ out)
{
    const int base_tok = blockIdx.x * 16;
    const int tid  = threadIdx.x;
    const int lane = tid & 31;
    const int wave = tid >> 5;
    const int lmod = lane & 15;
    const int lhi  = lane >> 4;          // 0 or 1

    __shared__ __align__(16) _Float16 sH1[16 * HID];   // layer-1 activations (f16)
    __shared__ __align__(16) float    sH2[16 * HID];   // layer-2 activations (f32)
    __shared__ float sPart[256][4];

    // ---- A fragment for layer 1 (16x32 f16, element e holds k = 16*(lane/16) + e) --------------
    v16h a1;
    {
        const _Float16* rp = X0 + (size_t)(base_tok + lmod) * X0K + (lhi ? 16 : 0);
        v8h lo = *(const v8h*)(rp);
        v8h hi = *(const v8h*)(rp + 8);
#pragma unroll
        for (int e = 0; e < 8; ++e) { a1[e] = lo[e]; a1[e + 8] = hi[e]; }
    }

    // ---- Layer 1: one WMMA per 16-wide N-tile ----------------------------
#pragma unroll
    for (int s = 0; s < 2; ++s) {
        const int nt = wave * 2 + s;
        v16h bf = *(const v16h*)(W1p + nt * 512 + lane * 16);
        v8f c = {};
        c = __builtin_amdgcn_wmma_f32_16x16x32_f16(false, a1, false, bf,
                                                   (short)0, c, false, false);
        const int n = nt * 16 + lmod;
        const float bias = b1[n];
        const int mbase = lhi * 8;
#pragma unroll
        for (int r = 0; r < 8; ++r) {
            float v = c[r] + bias;
            v = v > 0.f ? v : 0.f;
            sH1[(mbase + r) * HID + n] = (_Float16)v;
        }
    }
    __syncthreads();

    // ---- Layer 2: K=256 in 8 accumulating WMMA steps ---------------------
#pragma unroll
    for (int s = 0; s < 2; ++s) {
        const int nt = wave * 2 + s;
        v8f c = {};
#pragma unroll
        for (int ks = 0; ks < 8; ++ks) {
            v16h a;
            const _Float16* rp = &sH1[lmod * HID + ks * 32 + (lhi ? 16 : 0)];
            v8h lo = *(const v8h*)(rp);
            v8h hi = *(const v8h*)(rp + 8);
#pragma unroll
            for (int e = 0; e < 8; ++e) { a[e] = lo[e]; a[e + 8] = hi[e]; }
            v16h bf = *(const v16h*)(W2p + nt * 4096 + ks * 512 + lane * 16);
            c = __builtin_amdgcn_wmma_f32_16x16x32_f16(false, a, false, bf,
                                                       (short)0, c, false, false);
        }
        const int n = nt * 16 + lmod;
        const float bias = b2[n];
        const int mbase = lhi * 8;
#pragma unroll
        for (int r = 0; r < 8; ++r) {
            float v = c[r] + bias;
            v = v > 0.f ? v : 0.f;
            sH2[(mbase + r) * HID + n] = v;
        }
    }
    __syncthreads();

    // ---- Heads: 4 dots of length 256 per token (partial + reduce) --------
    {
        const int m = tid >> 4;   // token in tile
        const int p = tid & 15;   // 16-element chunk
        float p0 = 0.f, p1 = 0.f, p2 = 0.f, p3 = 0.f;
#pragma unroll
        for (int e = p * 16; e < p * 16 + 16; ++e) {
            float h = sH2[m * HID + e];
            p0 += h * Wmu[e * 2 + 0];
            p1 += h * Wmu[e * 2 + 1];
            p2 += h * Wls[e * 2 + 0];
            p3 += h * Wls[e * 2 + 1];
        }
        sPart[tid][0] = p0; sPart[tid][1] = p1;
        sPart[tid][2] = p2; sPart[tid][3] = p3;
    }
    __syncthreads();

    if (tid < 16) {
        const int m = tid;
        float mu0 = bmu[0], mu1 = bmu[1], ls0 = bls[0], ls1 = bls[1];
#pragma unroll
        for (int p = 0; p < 16; ++p) {
            mu0 += sPart[m * 16 + p][0]; mu1 += sPart[m * 16 + p][1];
            ls0 += sPart[m * 16 + p][2]; ls1 += sPart[m * 16 + p][3];
        }
        const int token = base_tok + m;
        const float mus[2] = { tanhf(mu0), tanhf(mu1) };
        const float lst[2] = { tanhf(ls0), tanhf(ls1) };
        float lp = 0.f;
#pragma unroll
        for (int jj = 0; jj < 2; ++jj) {
            float log_std = -20.0f + 11.0f * (lst[jj] + 1.0f);
            float std = __expf(log_std);
            float nz  = noise[(size_t)token * 2 + jj];
            float z   = mus[jj] + std * nz;
            float act = tanhf(z);
            float l = -0.5f * nz * nz - log_std - 0.9189385332046727f; // 0.5*ln(2pi)
            l -= logf(1.0f - act * act + 1e-7f);
            lp += l;
            out[(size_t)token * 2 + jj] = act;
        }
        out[(size_t)NTOK * 2 + token] = lp;
    }
}

// ---------------------------------------------------------------------------
// Host launcher
// ---------------------------------------------------------------------------
extern "C" void kernel_launch(void* const* d_in, const int* in_sizes, int n_in,
                              void* d_out, int out_size, void* d_ws, size_t ws_size,
                              hipStream_t stream)
{
    (void)in_sizes; (void)n_in; (void)out_size; (void)ws_size;

    const float* state = (const float*)d_in[0];
    const float* noise = (const float*)d_in[1];
    const float* Wq    = (const float*)d_in[2];
    const float* Wk    = (const float*)d_in[3];
    const float* Wv    = (const float*)d_in[4];
    const float* W1    = (const float*)d_in[5];
    const float* b1    = (const float*)d_in[6];
    const float* W2    = (const float*)d_in[7];
    const float* b2    = (const float*)d_in[8];
    const float* Wmu   = (const float*)d_in[9];
    const float* bmu   = (const float*)d_in[10];
    const float* Wls   = (const float*)d_in[11];
    const float* bls   = (const float*)d_in[12];
    float* out = (float*)d_out;

    char* ws = (char*)d_ws;
    _Float16* X0  = (_Float16*)ws;                                   // 65536*32 f16 = 4 MB
    _Float16* W1p = (_Float16*)(ws + (size_t)NTOK * X0K * 2);        // 8192 f16
    _Float16* W2p = (_Float16*)(ws + (size_t)NTOK * X0K * 2 + 8192 * 2); // 65536 f16

    // 1) pack weights into WMMA B-fragment order (f16)
    {
        const int total = 16 * 512 + 16 * 8 * 512;   // 73728
        prep_kernel<<<(total + 255) / 256, 256, 0, stream>>>(W1, W2, W1p, W2p);
    }
    // 2) attention / feature build (one block per token)
    attn_kernel<<<NTOK, 64, 0, stream>>>(state, Wq, Wk, Wv, X0);
    // 3) fused WMMA MLP + heads (16 tokens per block)
    mlp_kernel<<<NTOK / 16, 256, 0, stream>>>(X0, W1p, W2p, b1, b2,
                                              Wmu, bmu, Wls, bls, noise, out);
}
